// LutNet_79723182948578
// MI455X (gfx1250) — matrix operand order repaired and checked by reading, not compiled, not run on hardware
//
#include <hip/hip_runtime.h>
#include <hip/hip_bf16.h>
#include <math.h>

// ---------------------------------------------------------------------------
// LutNet on MI455X (gfx1250, wave32, WMMA, TDM)
//
// Pipeline (all on `stream`):
//   init        : zero stats + feature accumulators in ws
//   pool_stats  : 8x8 avg-pool img -> prior (8,3,128,128) + exact channel means
//   conv_wmma   : 10 layers x 3 encoders, implicit-GEMM via v_wmma_f32_16x16x32_f16
//   verts       : non-uniform LUT vertex curves from channel means
//   head        : classifier (hardswish/sigmoid) + LUT fusion per branch
//   apply       : per-pixel 3x AILUT trilinear + bright/mid/dark blend,
//                 fused LUTs staged to LDS via TENSOR_LOAD_TO_LDS (TDM) with
//                 per-lane global_load_async_to_lds_b32 as fallback
// ---------------------------------------------------------------------------

typedef __attribute__((ext_vector_type(16))) _Float16 v16h;
typedef __attribute__((ext_vector_type(8)))  float    v8f;

#if defined(__has_builtin)
#  if __has_builtin(__builtin_amdgcn_tensor_load_to_lds)
#    define HAVE_TDM 1
#  endif
#endif
#ifndef HAVE_TDM
#  define HAVE_TDM 0
#endif

#if HAVE_TDM
typedef __attribute__((ext_vector_type(4))) unsigned int u32x4;
typedef __attribute__((ext_vector_type(4))) int          i32x4;
typedef __attribute__((ext_vector_type(8))) int          i32x8;
#endif

// workspace layout (float offsets)
constexpr size_t WS_SUMS  = 0;                         // 3 channel sums
constexpr size_t WS_FEATS = 16;                        // 3 enc x 8 x 128
constexpr size_t WS_VERTS = 3104;                      // 3 branch x 3 ch x 17
constexpr size_t WS_FUSED = 3264;                      // 3 branch x 8 x 14739
constexpr size_t WS_PRIOR = 357376;                    // 8 x 3 x 16384
constexpr size_t WS_BUF0  = 750592;                    // 8 x 64 x 16384
constexpr size_t WS_BUF1  = WS_BUF0 + 8388608;
constexpr size_t WS_BUF2  = WS_BUF1 + 8388608;

// ---------------------------------------------------------------------------
__global__ void init_kernel(float* ws) {
    int i = blockIdx.x * 256 + threadIdx.x;
    if (i < (int)WS_VERTS) ws[i] = 0.0f;               // zero sums + feats
}

// ---------------------------------------------------------------------------
// 8x8 average pool (1024^2 -> 128^2) + per-channel sum of pooled values.
// mean(img channel) == sum(prior channel)/131072 exactly.
__global__ __launch_bounds__(256)
void pool_stats_kernel(const float* __restrict__ img,
                       float* __restrict__ prior,
                       float* __restrict__ sums) {
    __shared__ float red[256];
    const int tid = threadIdx.x;
    const int idx = blockIdx.x * 256 + tid;            // < 8*3*128*128
    const int px  = idx & 127;
    const int py  = (idx >> 7) & 127;
    const int bc  = idx >> 14;                         // b*3 + c
    const float* base = img + ((size_t)bc << 20) + (size_t)(py * 8) * 1024 + px * 8;
    float s = 0.0f;
#pragma unroll
    for (int r = 0; r < 8; ++r) {
        const float4* v = (const float4*)(base + (size_t)r * 1024);
        float4 a = v[0], b = v[1];
        s += a.x + a.y + a.z + a.w + b.x + b.y + b.z + b.w;
    }
    float m = s * (1.0f / 64.0f);
    prior[idx] = m;
    // whole block shares one (b,c): block-reduce then one atomic per block
    red[tid] = m;
    __syncthreads();
    for (int st = 128; st > 0; st >>= 1) {
        if (tid < st) red[tid] += red[tid + st];
        __syncthreads();
    }
    if (tid == 0) atomicAdd(&sums[bc % 3], red[0]);
}

// ---------------------------------------------------------------------------
// Implicit-GEMM conv with WMMA (f16 inputs, f32 accumulate).
//   A tile: 64 pixels x 32 K-slices (K = ci*KS*KS + ky*KS + kx, zero padded)
//   B tile: 32 K x COUT weights
// Block = 128 threads = 4 waves; wave w owns pixel rows [16w, 16w+16).
// Fragments are gathered from LDS in contiguous k order (k = 16*half + s) for A and B alike.
template<int CIN, int COUT, int KS, int PAD, bool LRELU, bool ADDRES, bool MEANOUT>
__global__ __launch_bounds__(128)
void conv_wmma_kernel(const float* __restrict__ in,    // (8, CIN, 128, 128)
                      const float* __restrict__ wgt,   // (COUT, CIN, KS, KS)
                      const float* __restrict__ bias,  // (COUT)
                      const float* __restrict__ res,   // residual or null
                      float* __restrict__ out)         // (8,COUT,128,128) or (8,COUT)
{
    constexpr int H = 128, W = 128, HW = H * W;
    constexpr int KTOT   = CIN * KS * KS;
    constexpr int NCHUNK = (KTOT + 31) / 32;
    constexpr int NT     = COUT / 16;

    __shared__ _Float16 As[64][32];
    __shared__ _Float16 Bs[32][COUT];

    const int tid  = threadIdx.x;
    const int lane = tid & 31;
    const int wv   = tid >> 5;
    const int half = lane >> 4;
    const int ln15 = lane & 15;
    const int b    = blockIdx.x >> 8;                  // image
    const int tile = blockIdx.x & 255;
    const int pixBase = tile * 64;
    const int m0   = wv * 16;

    v8f acc[NT];
#pragma unroll
    for (int nt = 0; nt < NT; ++nt) {
        float bv = bias[nt * 16 + ln15];
#pragma unroll
        for (int j = 0; j < 8; ++j) acc[nt][j] = bv;
    }

    for (int ch = 0; ch < NCHUNK; ++ch) {
        // ---- stage A: 64 x 32 activations (im2col on the fly, zero pad) ----
        for (int t = tid; t < 64 * 32; t += 128) {
            int m = t >> 5, kk = t & 31;
            int k = ch * 32 + kk;
            float v = 0.0f;
            if (k < KTOT) {
                int ci, dy, dx;
                if (KS == 1) { ci = k; dy = 0; dx = 0; }
                else {
                    ci = k / 9; int r2 = k - ci * 9;
                    int ky = r2 / 3; dy = ky - PAD; dx = (r2 - ky * 3) - PAD;
                }
                int p = pixBase + m;
                int y = (p >> 7) + dy, x = (p & 127) + dx;
                if (y >= 0 && y < H && x >= 0 && x < W)
                    v = in[((size_t)b * CIN + ci) * HW + y * W + x];
            }
            As[m][kk] = (_Float16)v;
        }
        // ---- stage B: 32 x COUT weights (w[n] is KTOT-contiguous) ----
        for (int t = tid; t < 32 * COUT; t += 128) {
            int kk = t / COUT, n = t - kk * COUT;
            int k = ch * 32 + kk;
            Bs[kk][n] = (_Float16)((k < KTOT) ? wgt[(size_t)n * KTOT + k] : 0.0f);
        }
        __syncthreads();

        // A fragment: lane half 0 -> K 0..15, half 1 -> K 16..31
        v16h a;
#pragma unroll
        for (int j = 0; j < 8; ++j) {
            a[j]     = As[m0 + ln15][half * 16 + j];
            a[8 + j] = As[m0 + ln15][half * 16 + 8 + j];
        }
#pragma unroll
        for (int nt = 0; nt < NT; ++nt) {
            // B fragment: lanes 0-15 K 0..15, lanes 16-31 K 16..31, col = lane&15
            v16h bf;
#pragma unroll
            for (int j = 0; j < 16; ++j) bf[j] = Bs[half * 16 + j][nt * 16 + ln15];
            acc[nt] = __builtin_amdgcn_wmma_f32_16x16x32_f16(
                false, a, false, bf, (short)0, acc[nt], false, false);
        }
        __syncthreads();
    }

    // epilogue: C/D layout => element j is row M = j + 8*half, col = lane&15
#pragma unroll
    for (int nt = 0; nt < NT; ++nt) {
        int n = nt * 16 + ln15;
        if (MEANOUT) {
            float s = 0.0f;
#pragma unroll
            for (int j = 0; j < 8; ++j) s += acc[nt][j];
            s += __shfl_xor(s, 16, 32);                // combine M halves
            if (half == 0)
                atomicAdd(&out[(size_t)b * COUT + n], s * (1.0f / 16384.0f));
        } else {
#pragma unroll
            for (int j = 0; j < 8; ++j) {
                int p = pixBase + m0 + half * 8 + j;
                float v = acc[nt][j];
                if (LRELU)  v = (v >= 0.0f) ? v : 0.2f * v;
                if (ADDRES) v += res[((size_t)b * COUT + n) * HW + p];
                out[((size_t)b * COUT + n) * HW + p] = v;
            }
        }
    }
}

// ---------------------------------------------------------------------------
// Non-uniform vertex curves from exact channel means.
__global__ void verts_kernel(const float* __restrict__ sums, float* __restrict__ verts) {
    int t = threadIdx.x;
    if (t >= 153) return;                              // 3 branch x 3 ch x 17
    int e = t / 51, r = t - e * 51, c = r / 17, i = r - c * 17;
    float u   = (float)i * (1.0f / 16.0f);
    float avg = sums[c] * (1.0f / 131072.0f);
    float v;
    if (e == 1) {                                      // mid curve
        const float PI3 = 3.0f * 3.14159265358979f;
        v = (PI3 * u - __cosf(PI3 * u) + 1.0f) / (PI3 + 2.0f);
    } else {
        float ex = (e == 0) ? 1.0f / (1.4f + 0.8f * avg)   // bright: u^(1/e)
                            : (2.2f - 0.8f * avg);          // dark:   u^e
        v = (u > 0.0f) ? __powf(u, ex) : 0.0f;
    }
    verts[t] = v;
}

// ---------------------------------------------------------------------------
// Classifier head + LUT fusion for one branch. Block per batch image.
__global__ __launch_bounds__(128)
void head_kernel(const float* __restrict__ feat,   // (8,128) for this branch
                 const float* __restrict__ w0, const float* __restrict__ b0,
                 const float* __restrict__ w1, const float* __restrict__ b1,
                 const float* __restrict__ luts,   // (5,3,17,17,17)
                 float* __restrict__ fused)        // (8, 3*4913)
{
    __shared__ float h[128];
    __shared__ float red[128];
    __shared__ float wl[5];
    const int bi  = blockIdx.x;
    const int tid = threadIdx.x;
    const float* f = feat + bi * 128;

    float acc = b0[tid];
    for (int k = 0; k < 128; ++k) acc += f[k] * w0[tid * 128 + k];
    float t3 = fminf(fmaxf(acc + 3.0f, 0.0f), 6.0f);   // hardswish
    h[tid] = acc * t3 * (1.0f / 6.0f);
    __syncthreads();

    for (int l = 0; l < 5; ++l) {
        red[tid] = h[tid] * w1[l * 128 + tid];
        __syncthreads();
        for (int st = 64; st > 0; st >>= 1) {
            if (tid < st) red[tid] += red[tid + st];
            __syncthreads();
        }
        if (tid == 0) wl[l] = 1.0f / (1.0f + __expf(-(red[0] + b1[l])));
        __syncthreads();
    }
    for (int i = tid; i < 14739; i += 128) {           // 3*17^3
        float s = 0.0f;
#pragma unroll
        for (int l = 0; l < 5; ++l) s += wl[l] * luts[l * 14739 + i];
        fused[(size_t)bi * 14739 + i] = s;
    }
}

// ---------------------------------------------------------------------------
// Per-pixel AILUT + blend. The three fused LUTs of this image (173 KB) are
// staged into LDS, exploiting the 320 KB/WGP LDS. Preferred path: Tensor Data
// Mover — one wave issues 3 descriptor-driven DMA tiles (contiguous 14739 f32
// rows), tracked by TENSORcnt, freeing the other waves' issue slots. Fallback:
// per-lane async-to-LDS loads (ASYNCcnt). 128 blocks per image x 8192 pixels.
__global__ __launch_bounds__(256)
void apply_kernel(const float* __restrict__ img,
                  const float* __restrict__ ws,
                  float* __restrict__ out)
{
    extern __shared__ float smem[];                    // [3][3][4913] + [3][3][17]
    constexpr int LUTSZ = 3 * 3 * 4913;                // 44217
    constexpr int HW2   = 1024 * 1024;
    const int tid = threadIdx.x;
    const int bi  = blockIdx.x >> 7;
    const int blk = blockIdx.x & 127;

    const float* fused = ws + WS_FUSED;

#if HAVE_TDM
    if (tid < 32) {                                    // wave 0 drives the TDM
#pragma unroll
        for (int e = 0; e < 3; ++e) {
            const float* src = fused + ((size_t)e * 8 + bi) * 14739;
            unsigned long long ga = (unsigned long long)(const void*)src;
            // D# group 0: count=1 | lds_addr | 57-bit global_addr | type=2
            u32x4 g0;
            g0[0] = 1u;
            g0[1] = (unsigned)(e * 14739 * 4);
            g0[2] = (unsigned)(ga & 0xFFFFFFFFull);
            g0[3] = (unsigned)((ga >> 32) & 0x1FFFFFFull) | (2u << 30);
            // D# group 1: data_size=4B, 1-D tile: tensor_dim0 = tile_dim0 = 14739
            i32x8 g1;
            g1[0] = (int)(2u << 16);                   // data_size=2 (4 bytes)
            g1[1] = (int)((14739u & 0xFFFFu) << 16);   // tensor_dim0[15:0]
            g1[2] = (int)(((14739u >> 16) & 0xFFFFu) | (1u << 16)); // dim0 hi | tensor_dim1=1
            g1[3] = (int)(14739u << 16);               // tile_dim0
            g1[4] = 1;                                 // tile_dim1=1, tile_dim2=0
            g1[5] = 14739;                             // tensor_dim0_stride
            g1[6] = 0;
            g1[7] = 0;
            i32x4 g2 = {0, 0, 0, 0};
            i32x4 g3 = {0, 0, 0, 0};
#if __clang_major__ >= 23
            i32x8 g4 = {0, 0, 0, 0, 0, 0, 0, 0};
            __builtin_amdgcn_tensor_load_to_lds(g0, g1, g2, g3, g4, 0);
#else
            __builtin_amdgcn_tensor_load_to_lds(g0, g1, g2, g3, 0);
#endif
        }
        __builtin_amdgcn_s_wait_tensorcnt(0);
    }
#else
    // per-lane async copy fallback (dynamic LDS starts at offset 0 here)
#pragma unroll
    for (int e = 0; e < 3; ++e) {
        const float* src = fused + ((size_t)e * 8 + bi) * 14739;
        for (int i = tid; i < 14739; i += 256) {
            unsigned lofs = (unsigned)((e * 14739 + i) * 4);
            unsigned long long ga = (unsigned long long)(const void*)(src + i);
            asm volatile("global_load_async_to_lds_b32 %0, %1, off"
                         :: "v"(lofs), "v"(ga) : "memory");
        }
    }
    asm volatile("s_wait_asynccnt 0" ::: "memory");
#endif

    // vertices via ordinary LDS stores
    const float* verts = ws + WS_VERTS;
    for (int i = tid; i < 153; i += 256) smem[LUTSZ + i] = verts[i];
    __syncthreads();

    const float* vB = smem + LUTSZ;
    const float* imR = img + (size_t)bi * 3 * HW2;
    float*       oR  = out + (size_t)bi * 3 * HW2;

    for (int it = 0; it < 32; ++it) {
        int p = blk * 8192 + it * 256 + tid;
        float px[3] = { imR[p], imR[HW2 + p], imR[2 * HW2 + p] };
        __builtin_prefetch(&imR[p + 256], 0, 0);       // global_prefetch_b8
        float o[3] = { 0.0f, 0.0f, 0.0f };
#pragma unroll
        for (int e = 0; e < 3; ++e) {
            int   idx[3]; float fr[3];
#pragma unroll
            for (int c = 0; c < 3; ++c) {
                const float* v = vB + (e * 3 + c) * 17;
                float x = px[c];
                int i0 = 0;
#pragma unroll
                for (int t = 1; t <= 15; ++t) i0 = (x >= v[t]) ? t : i0;
                float lo = v[i0], hi = v[i0 + 1];
                fr[c]  = fminf(fmaxf((x - lo) / (hi - lo + 1e-10f), 0.0f), 1.0f);
                idx[c] = i0;
            }
            float f0 = fr[0], f1 = fr[1], f2 = fr[2];
            float w000 = (1 - f0) * (1 - f1) * (1 - f2), w001 = (1 - f0) * (1 - f1) * f2;
            float w010 = (1 - f0) * f1 * (1 - f2),       w011 = (1 - f0) * f1 * f2;
            float w100 = f0 * (1 - f1) * (1 - f2),       w101 = f0 * (1 - f1) * f2;
            float w110 = f0 * f1 * (1 - f2),             w111 = f0 * f1 * f2;
            int base = idx[0] * 289 + idx[1] * 17 + idx[2];
#pragma unroll
            for (int co = 0; co < 3; ++co) {
                const float* L = smem + (e * 3 + co) * 4913 + base;
                float s = w000 * L[0]   + w001 * L[1]
                        + w010 * L[17]  + w011 * L[18]
                        + w100 * L[289] + w101 * L[290]
                        + w110 * L[306] + w111 * L[307];
                float xv = px[co], m;
                if (e == 0)      m = fminf(fmaxf((xv - 0.55f) * (1.0f / 0.45f), 0.0f), 1.0f);
                else if (e == 2) m = fminf(fmaxf((0.45f - xv) * (1.0f / 0.45f), 0.0f), 1.0f);
                else {
                    float br = fminf(fmaxf((xv - 0.55f) * (1.0f / 0.45f), 0.0f), 1.0f);
                    float dk = fminf(fmaxf((0.45f - xv) * (1.0f / 0.45f), 0.0f), 1.0f);
                    m = fminf(fmaxf(1.0f - br - dk, 0.0f), 1.0f);
                }
                o[co] += m * s;
            }
        }
        oR[p] = o[0]; oR[HW2 + p] = o[1]; oR[2 * HW2 + p] = o[2];
    }
}

// ---------------------------------------------------------------------------
extern "C" void kernel_launch(void* const* d_in, const int* in_sizes, int n_in,
                              void* d_out, int out_size, void* d_ws, size_t ws_size,
                              hipStream_t stream)
{
    (void)in_sizes; (void)n_in; (void)out_size; (void)ws_size;
    auto P = [&](int i) { return (const float*)d_in[i]; };
    float* ws  = (float*)d_ws;
    const float* img = P(0);
    float* out = (float*)d_out;

    float* sums  = ws + WS_SUMS;
    float* feats = ws + WS_FEATS;
    float* verts = ws + WS_VERTS;
    float* fused = ws + WS_FUSED;
    float* prior = ws + WS_PRIOR;
    float* buf0  = ws + WS_BUF0;
    float* buf1  = ws + WS_BUF1;
    float* buf2  = ws + WS_BUF2;

    init_kernel<<<16, 256, 0, stream>>>(ws);
    pool_stats_kernel<<<1536, 256, 0, stream>>>(img, prior, sums);

    // jax-pytree (alphabetical) flat order: enc_b @13, enc_d @33, enc_m @53.
    // feat slots in branch order: 0=bright(enc_b), 1=mid(enc_m), 2=dark(enc_d)
    const int encBase[3] = { 13, 53, 33 };
    for (int s = 0; s < 3; ++s) {
        const int B = encBase[s];
        float* feat = feats + (size_t)s * 8 * 128;
        const float *c0b = P(B+0),  *c0w = P(B+1),  *c1b = P(B+2),  *c1w = P(B+3),
                    *c2b = P(B+4),  *c2w = P(B+5),  *c3b = P(B+6),  *c3w = P(B+7),
                    *r0ab= P(B+8),  *r0aw= P(B+9),  *r0bb= P(B+10), *r0bw= P(B+11),
                    *r1ab= P(B+12), *r1aw= P(B+13), *r1bb= P(B+14), *r1bw= P(B+15),
                    *r2ab= P(B+16), *r2aw= P(B+17), *r2bb= P(B+18), *r2bw= P(B+19);

        conv_wmma_kernel< 3, 16,1,0,false,false,false><<<2048,128,0,stream>>>(prior, c0w, c0b, nullptr, buf0);
        conv_wmma_kernel<16, 16,3,1,true, false,false><<<2048,128,0,stream>>>(buf0, r0aw, r0ab, nullptr, buf1);
        conv_wmma_kernel<16, 16,3,1,true, true, false><<<2048,128,0,stream>>>(buf1, r0bw, r0bb, buf0,   buf2);
        conv_wmma_kernel<16, 32,1,0,false,false,false><<<2048,128,0,stream>>>(buf2, c1w,  c1b,  nullptr, buf0);
        conv_wmma_kernel<32, 32,3,1,true, false,false><<<2048,128,0,stream>>>(buf0, r1aw, r1ab, nullptr, buf1);
        conv_wmma_kernel<32, 32,3,1,true, true, false><<<2048,128,0,stream>>>(buf1, r1bw, r1bb, buf0,   buf2);
        conv_wmma_kernel<32, 64,1,0,false,false,false><<<2048,128,0,stream>>>(buf2, c2w,  c2b,  nullptr, buf0);
        conv_wmma_kernel<64, 64,3,1,true, false,false><<<2048,128,0,stream>>>(buf0, r2aw, r2ab, nullptr, buf1);
        conv_wmma_kernel<64, 64,3,1,true, true, false><<<2048,128,0,stream>>>(buf1, r2bw, r2bb, buf0,   buf2);
        conv_wmma_kernel<64,128,1,0,false,false,true ><<<2048,128,0,stream>>>(buf2, c3w,  c3b,  nullptr, feat);
    }

    verts_kernel<<<1, 192, 0, stream>>>(sums, verts);

    // cls_b: b0=1,b1=2,w0=3,w1=4 ; cls_m: b0=9,b1=10,w0=11,w1=12 ; luts_b=73, luts_m=75
    // branch 0 (bright): feat_b, cls_b, luts_b
    head_kernel<<<8,128,0,stream>>>(feats + 0*1024, P(3),  P(1), P(4),  P(2),  P(73), fused + (size_t)0*8*14739);
    // branch 1 (mid):    feat_m, cls_m, luts_m
    head_kernel<<<8,128,0,stream>>>(feats + 1*1024, P(11), P(9), P(12), P(10), P(75), fused + (size_t)1*8*14739);
    // branch 2 (dark):   feat_d with cls_b / luts_b (reference quirk)
    head_kernel<<<8,128,0,stream>>>(feats + 2*1024, P(3),  P(1), P(4),  P(2),  P(73), fused + (size_t)2*8*14739);

    const size_t smem = (3 * 3 * 4913 + 3 * 3 * 17) * sizeof(float);  // ~173.5 KB of 320 KB/WGP
    apply_kernel<<<1024, 256, smem, stream>>>(img, ws, out);
}
